// NystromAttention_19980187861862
// MI455X (gfx1250) — compile-verified, matrix operand order checked by reading, not run on hardware
//
#include <hip/hip_runtime.h>
#include <hip/hip_bf16.h>

// ---------------------------------------------------------------------------
// Nystrom attention for MI455X (gfx1250).
// bf16 WMMA GEMM (f32 accumulate) with LDS double-buffering fed by
// global_load_async_to_lds_b128 (ASYNCcnt-tracked async copy engine).
// ---------------------------------------------------------------------------

typedef __attribute__((ext_vector_type(16))) __bf16 v16bf;
typedef __attribute__((ext_vector_type(8)))  __bf16 v8bf;
typedef __attribute__((ext_vector_type(8)))  float  v8f;

#define LDA 40          // padded LDS row stride in bf16 (80B -> bank-conflict-free b128 reads)
#define LDSBUF (64 * LDA)   // elements per buffer (one 64 x 32 tile, padded)

union frag_u { v16bf v; v8bf h[2]; };

__device__ __forceinline__ void async_b128(unsigned ldsOff, unsigned long long gaddr) {
    // LDS[wave_base + ldsOff .. +15] = MEM[gaddr .. +15], tracked by ASYNCcnt
    asm volatile("global_load_async_to_lds_b128 %0, %1, off"
                 :: "v"(ldsOff), "v"(gaddr) : "memory");
}
__device__ __forceinline__ void wait_async0() {
    asm volatile("s_wait_asynccnt 0x0" ::: "memory");
}

// ---------------------------------------------------------------------------
// C[z][M,N] = A[z][M,K] * Bt[z][N,K]^T (+ bias[n])
//   A  : bf16 row-major M x K;  Bt : bf16 K-major N x K
//   Cf : optional f32 out, Cbf : optional bf16 out (same element stride sC)
// Block = 128 threads (4 waves). Macro tile 64(M) x 64(N); 32x32 per wave.
// Requires M%64==0, N%64==0, K%32==0 (true for every call below).
// ---------------------------------------------------------------------------
__global__ __launch_bounds__(128)
void wmma_gemm_bf16(const __bf16* __restrict__ A,
                    const __bf16* __restrict__ Bt,
                    const float*  __restrict__ bias,
                    float*        __restrict__ Cf,
                    __bf16*       __restrict__ Cbf,
                    int M, int N, int K,
                    long sA, long sB, long sC) {
    __shared__ __align__(16) __bf16 As[2][LDSBUF];
    __shared__ __align__(16) __bf16 Bs[2][LDSBUF];

    const int z = blockIdx.z;
    A  += (long)z * sA;
    Bt += (long)z * sB;

    const long mBase = (long)blockIdx.y * 64;
    const long nBase = (long)blockIdx.x * 64;

    const int t    = threadIdx.x;          // 0..127
    const int lane = t & 31;
    const int w    = t >> 5;               // wave id 0..3
    const int half = lane >> 4;            // K-half selector for WMMA fragments
    const int idx  = lane & 15;
    const int mo   = (w >> 1) * 32;        // wave's 32x32 sub-tile inside 64x64
    const int no   = (w & 1) * 32;

    // ---- async staging setup: 2 b128 pieces of A and 2 of B per thread/chunk
    // piece P (0..255): row r = P>>2 (0..63), 16B-piece j = P&3 within 64B row
    const int P0 = t, P1 = t + 128;
    const int rA0 = P0 >> 2, jA0 = P0 & 3;
    const int rA1 = P1 >> 2, jA1 = P1 & 3;

    unsigned long long gA0 = (unsigned long long)(const void*)(A  + (mBase + rA0) * (long)K + jA0 * 8);
    unsigned long long gA1 = (unsigned long long)(const void*)(A  + (mBase + rA1) * (long)K + jA1 * 8);
    unsigned long long gB0 = (unsigned long long)(const void*)(Bt + (nBase + rA0) * (long)K + jA0 * 8);
    unsigned long long gB1 = (unsigned long long)(const void*)(Bt + (nBase + rA1) * (long)K + jA1 * 8);

    const unsigned ldsAbase = (unsigned)(uintptr_t)(&As[0][0]);
    const unsigned ldsBbase = (unsigned)(uintptr_t)(&Bs[0][0]);
    const unsigned lA0 = (unsigned)((rA0 * LDA + jA0 * 8) * 2);
    const unsigned lA1 = (unsigned)((rA1 * LDA + jA1 * 8) * 2);
    const unsigned bufBytes = LDSBUF * 2;

    auto issue = [&](int buf) {
        const unsigned ao = ldsAbase + (unsigned)buf * bufBytes;
        const unsigned bo = ldsBbase + (unsigned)buf * bufBytes;
        async_b128(ao + lA0, gA0);
        async_b128(ao + lA1, gA1);
        async_b128(bo + lA0, gB0);
        async_b128(bo + lA1, gB1);
        gA0 += 64; gA1 += 64; gB0 += 64; gB1 += 64;   // advance one 32-elem chunk
    };

    v8f acc00 = {}, acc01 = {}, acc10 = {}, acc11 = {};

    const int nChunks = K >> 5;
    issue(0);                                          // prologue: chunk 0

    for (int c = 0; c < nChunks; ++c) {
        wait_async0();          // this wave's chunk-c pieces are in LDS
        __syncthreads();        // all waves' pieces in; prev buffer reads retired
        if (c + 1 < nChunks) issue((c + 1) & 1);       // overlap copy with math

        const __bf16* Ab = &As[c & 1][0];
        const __bf16* Bb = &Bs[c & 1][0];

        const __bf16* ar0 = Ab + (mo + idx) * LDA + 8 * half;
        const __bf16* br0 = Bb + (no + idx) * LDA + 8 * half;

        frag_u a0, a1, b0, b1;
        a0.h[0] = *(const v8bf*)(ar0);
        a0.h[1] = *(const v8bf*)(ar0 + 16);
        a1.h[0] = *(const v8bf*)(ar0 + 16 * LDA);
        a1.h[1] = *(const v8bf*)(ar0 + 16 * LDA + 16);
        b0.h[0] = *(const v8bf*)(br0);
        b0.h[1] = *(const v8bf*)(br0 + 16);
        b1.h[0] = *(const v8bf*)(br0 + 16 * LDA);
        b1.h[1] = *(const v8bf*)(br0 + 16 * LDA + 16);

        acc00 = __builtin_amdgcn_wmma_f32_16x16x32_bf16(false, a0.v, false, b0.v, (short)0, acc00, false, false);
        acc01 = __builtin_amdgcn_wmma_f32_16x16x32_bf16(false, a0.v, false, b1.v, (short)0, acc01, false, false);
        acc10 = __builtin_amdgcn_wmma_f32_16x16x32_bf16(false, a1.v, false, b0.v, (short)0, acc10, false, false);
        acc11 = __builtin_amdgcn_wmma_f32_16x16x32_bf16(false, a1.v, false, b1.v, (short)0, acc11, false, false);
    }

    // ---- epilogue: bias + store (C/D layout: VGPR i -> row half*8+i, col idx)
    const long col0 = nBase + no + idx;
    const long col1 = col0 + 16;
    const float bv0 = bias ? bias[col0] : 0.0f;
    const float bv1 = bias ? bias[col1] : 0.0f;
    const long cz = (long)z * sC;

    #pragma unroll
    for (int i = 0; i < 8; ++i) {
        const long row0 = mBase + mo + half * 8 + i;
        const long row1 = row0 + 16;
        const float v00 = acc00[i] + bv0, v01 = acc01[i] + bv1;
        const float v10 = acc10[i] + bv0, v11 = acc11[i] + bv1;
        if (Cf) {
            Cf[cz + row0 * (long)N + col0] = v00;
            Cf[cz + row0 * (long)N + col1] = v01;
            Cf[cz + row1 * (long)N + col0] = v10;
            Cf[cz + row1 * (long)N + col1] = v11;
        }
        if (Cbf) {
            Cbf[cz + row0 * (long)N + col0] = (__bf16)v00;
            Cbf[cz + row0 * (long)N + col1] = (__bf16)v01;
            Cbf[cz + row1 * (long)N + col0] = (__bf16)v10;
            Cbf[cz + row1 * (long)N + col1] = (__bf16)v11;
        }
    }
}

// ---------------------------------------------------------------------------
// Elementwise cast f32 -> bf16 (grid-stride).
// ---------------------------------------------------------------------------
__global__ void cast_f32_bf16(const float* __restrict__ in,
                              __bf16* __restrict__ out, long n) {
    long i = (long)blockIdx.x * blockDim.x + threadIdx.x;
    long stride = (long)gridDim.x * blockDim.x;
    for (; i < n; i += stride) out[i] = (__bf16)in[i];
}

// ---------------------------------------------------------------------------
// LDS-tiled transpose + optional elementwise multiply + cast to bf16.
//   X[z] f32 R x C  ->  out[z] bf16 C x R ; optional Ymul multiplied in.
// ---------------------------------------------------------------------------
__global__ __launch_bounds__(256)
void transpose_mul_cast(const float* __restrict__ X,
                        const float* __restrict__ Ymul,
                        __bf16* __restrict__ out,
                        int R, int C, long sIn, long sOut) {
    __shared__ float tile[16][17];
    const int z = blockIdx.z;
    X += (long)z * sIn;
    if (Ymul) Ymul += (long)z * sIn;
    out += (long)z * sOut;

    const int tx = threadIdx.x, ty = threadIdx.y;
    const long r = (long)blockIdx.y * 16 + ty;
    const long c = (long)blockIdx.x * 16 + tx;

    float v = X[r * C + c];
    if (Ymul) v *= Ymul[r * C + c];
    tile[ty][tx] = v;
    __syncthreads();

    const long oc  = (long)blockIdx.x * 16 + ty;
    const long orr = (long)blockIdx.y * 16 + tx;
    out[oc * R + orr] = (__bf16)tile[tx][ty];
}

// ---------------------------------------------------------------------------
// Row softmax over D=1024, in-place f32 + bf16 copy. One block (256 thr)/row.
// ---------------------------------------------------------------------------
__global__ __launch_bounds__(256)
void softmax_rows(float* __restrict__ X, __bf16* __restrict__ outBF, int D) {
    __shared__ float red[256];
    const long row = blockIdx.x;
    float* x = X + row * (long)D;
    const int tid = threadIdx.x;

    float m = -3.402823466e38f;
    for (int j = tid; j < D; j += 256) m = fmaxf(m, x[j]);
    red[tid] = m; __syncthreads();
    for (int s = 128; s > 0; s >>= 1) {
        if (tid < s) red[tid] = fmaxf(red[tid], red[tid + s]);
        __syncthreads();
    }
    m = red[0]; __syncthreads();

    float e[4];
    float sum = 0.0f;
    int cnt = 0;
    for (int j = tid; j < D; j += 256) {
        float tv = __expf(x[j] - m);
        e[cnt++] = tv;
        sum += tv;
    }
    red[tid] = sum; __syncthreads();
    for (int s = 128; s > 0; s >>= 1) {
        if (tid < s) red[tid] += red[tid + s];
        __syncthreads();
    }
    const float inv = 1.0f / red[0];

    cnt = 0;
    for (int j = tid; j < D; j += 256) {
        float tv = e[cnt++] * inv;
        x[j] = tv;
        outBF[row * (long)D + j] = (__bf16)tv;
    }
}

// ---------------------------------------------------------------------------
// Landmark mean-pool: KL[b,l,d] = mean over (S/L) rows; bf16 out.
// ---------------------------------------------------------------------------
__global__ __launch_bounds__(256)
void landmark_pool(const float* __restrict__ Ksm, __bf16* __restrict__ KL,
                   int S, int D, int L) {
    const int bl = blockIdx.x;
    const int rows = S / L;
    const float* base = Ksm + (long)bl * rows * D;
    const float invRows = 1.0f / (float)rows;
    for (int d = threadIdx.x; d < D; d += blockDim.x) {
        float s = 0.0f;
        for (int r = 0; r < rows; ++r) s += base[(long)r * D + d];
        KL[(long)bl * D + d] = (__bf16)(s * invRows);
    }
}

// ---------------------------------------------------------------------------
// Host-side launcher
// ---------------------------------------------------------------------------
extern "C" void kernel_launch(void* const* d_in, const int* in_sizes, int n_in,
                              void* d_out, int out_size, void* d_ws, size_t ws_size,
                              hipStream_t stream) {
    (void)in_sizes; (void)n_in; (void)out_size; (void)ws_size;

    constexpr int B = 4, S = 4096, D = 1024, L = 64;
    constexpr long BS = (long)B * S;          // 16384 rows

    const float* hidden = (const float*)d_in[0];
    const float* Wq = (const float*)d_in[1];
    const float* bq = (const float*)d_in[2];
    const float* Wk = (const float*)d_in[3];
    const float* bk = (const float*)d_in[4];
    const float* Wv = (const float*)d_in[5];
    const float* bv = (const float*)d_in[6];
    float* out = (float*)d_out;

    // ---- workspace carve-out (256B aligned) ----
    size_t off = 0;
    char* base = (char*)d_ws;
    auto alloc = [&](size_t bytes) -> void* {
        void* p = base + off;
        off += (bytes + 255) & ~(size_t)255;
        return p;
    };
    __bf16* hbf   = (__bf16*)alloc(BS * D * 2);
    __bf16* wT    = (__bf16*)alloc(3L * D * D * 2);
    float*  qf    = (float*) alloc(BS * D * 4);
    float*  kf    = (float*) alloc(BS * D * 4);
    float*  vf    = (float*) alloc(BS * D * 4);
    __bf16* qbf   = (__bf16*)alloc(BS * D * 2);
    __bf16* kbf   = (__bf16*)alloc(BS * D * 2);
    __bf16* klbf  = (__bf16*)alloc((long)B * L * D * 2);
    __bf16* k1bf  = (__bf16*)alloc((long)B * S * L * 2);
    __bf16* k2bf  = (__bf16*)alloc((long)B * L * S * 2);
    __bf16* k3t   = (__bf16*)alloc((long)B * D * S * 2);
    float*  midf  = (float*) alloc((long)B * L * D * 4);
    __bf16* midT  = (__bf16*)alloc((long)B * D * L * 2);

    // 1) casts / weight transposes
    cast_f32_bf16<<<4096, 256, 0, stream>>>(hidden, hbf, BS * (long)D);
    {
        dim3 blk(16, 16), grd(D / 16, D / 16, 1);
        transpose_mul_cast<<<grd, blk, 0, stream>>>(Wq, nullptr, wT + 0L * D * D, D, D, 0, 0);
        transpose_mul_cast<<<grd, blk, 0, stream>>>(Wk, nullptr, wT + 1L * D * D, D, D, 0, 0);
        transpose_mul_cast<<<grd, blk, 0, stream>>>(Wv, nullptr, wT + 2L * D * D, D, D, 0, 0);
    }

    // 2) QKV projections: [16384,1024] = hbf @ W (+bias)
    {
        dim3 grd(D / 64, BS / 64, 1);
        wmma_gemm_bf16<<<grd, 128, 0, stream>>>(hbf, wT + 0L * D * D, bq, qf, nullptr,
                                                (int)BS, D, D, 0, 0, 0);
        wmma_gemm_bf16<<<grd, 128, 0, stream>>>(hbf, wT + 1L * D * D, bk, kf, nullptr,
                                                (int)BS, D, D, 0, 0, 0);
        wmma_gemm_bf16<<<grd, 128, 0, stream>>>(hbf, wT + 2L * D * D, bv, vf, nullptr,
                                                (int)BS, D, D, 0, 0, 0);
    }

    // 3) feature-dim softmax on q and k
    softmax_rows<<<(int)BS, 256, 0, stream>>>(qf, qbf, D);
    softmax_rows<<<(int)BS, 256, 0, stream>>>(kf, kbf, D);

    // 4) landmark pooling + kernel_3^T = (k*v)^T (bf16, K-major over S)
    landmark_pool<<<B * L, 256, 0, stream>>>(kf, klbf, S, D, L);
    {
        dim3 blk(16, 16), grd(D / 16, S / 16, B);
        transpose_mul_cast<<<grd, blk, 0, stream>>>(kf, vf, k3t, S, D,
                                                    (long)S * D, (long)S * D);
    }

    // 5) kernel_1 = q @ kl^T : [B, S, L]
    {
        dim3 grd(L / 64, S / 64, B);
        wmma_gemm_bf16<<<grd, 128, 0, stream>>>(qbf, klbf, nullptr, nullptr, k1bf,
                                                S, L, D,
                                                (long)S * D, (long)L * D, (long)S * L);
    }
    // 6) kernel_2 = kl @ k^T : [B, L, S]
    {
        dim3 grd(S / 64, L / 64, B);
        wmma_gemm_bf16<<<grd, 128, 0, stream>>>(klbf, kbf, nullptr, nullptr, k2bf,
                                                L, S, D,
                                                (long)L * D, (long)S * D, (long)L * S);
    }
    // 7) mid = kernel_2 @ kernel_3 : [B, L, D]
    {
        dim3 grd(D / 64, L / 64, B);
        wmma_gemm_bf16<<<grd, 128, 0, stream>>>(k2bf, k3t, nullptr, midf, nullptr,
                                                L, D, S,
                                                (long)L * S, (long)D * S, (long)L * D);
    }
    // 8) mid^T (bf16, K-major over L)
    {
        dim3 blk(16, 16), grd(D / 16, L / 16, B);
        transpose_mul_cast<<<grd, blk, 0, stream>>>(midf, nullptr, midT, L, D,
                                                    (long)L * D, (long)D * L);
    }
    // 9) output = kernel_1 @ mid : [B, S, D], K = 64, straight to d_out
    {
        dim3 grd(D / 64, S / 64, B);
        wmma_gemm_bf16<<<grd, 128, 0, stream>>>(k1bf, midT, nullptr, out, nullptr,
                                                S, D, L,
                                                (long)S * L, (long)D * L, (long)S * D);
    }
}
